// GEmbedNet_88064009437952
// MI455X (gfx1250) — matrix operand order repaired and checked by reading, compiled, not run on hardware
//
#include <hip/hip_runtime.h>
#include <hip/hip_bf16.h>
#include <stdint.h>

#define N_NODES 50000
#define N_EDGES 800000
#define IN_DIM  89
#define KP1     96     // IN_DIM padded to multiple of 32
#define HID     128

#if defined(__HIP_DEVICE_COMPILE__) && \
    __has_builtin(__builtin_amdgcn_global_load_async_to_lds_b128) && \
    __has_builtin(__builtin_amdgcn_s_wait_asynccnt)
#define HAVE_ASYNC_LDS 1
#else
#define HAVE_ASYNC_LDS 0
#endif

typedef __attribute__((ext_vector_type(16))) __bf16 v16bf;
typedef __attribute__((ext_vector_type(8)))  float  v8f;
typedef int v4i __attribute__((vector_size(16)));
typedef __attribute__((address_space(1))) v4i* gptr_v4i;
typedef __attribute__((address_space(3))) v4i* lptr_v4i;

union FragU { uint4 u[2]; v16bf v; };

__device__ __forceinline__ unsigned short f2bf(float f) {
  union { float f; uint32_t u; } x; x.f = f;
  uint32_t u = x.u;
  u += 0x7fffu + ((u >> 16) & 1u);   // round-to-nearest-even
  return (unsigned short)(u >> 16);
}

// ---------------- degree / norm ----------------
__global__ void k_init_deg(float* deg, int n) {
  int i = blockIdx.x * blockDim.x + threadIdx.x;
  if (i < n) deg[i] = 1.0f;          // self loop
}
__global__ void k_deg_accum(const int* __restrict__ dst, float* __restrict__ deg, int e) {
  int i = blockIdx.x * blockDim.x + threadIdx.x;
  if (i < e) atomicAdd(&deg[dst[i]], 1.0f);
}
__global__ void k_rsqrt(const float* __restrict__ deg, float* __restrict__ dis, int n) {
  int i = blockIdx.x * blockDim.x + threadIdx.x;
  if (i < n) dis[i] = rsqrtf(deg[i]);
}

// ---------------- f32 -> bf16 convert with padding ----------------
__global__ void k_pad_cols_bf16(const float* __restrict__ in, unsigned short* __restrict__ out,
                                int rows, int incols, int outcols) {
  int t = blockIdx.x * blockDim.x + threadIdx.x;
  if (t >= rows * outcols) return;
  int r = t / outcols, c = t - r * outcols;
  float v = (c < incols) ? in[(size_t)r * incols + c] : 0.0f;
  out[t] = f2bf(v);
}
__global__ void k_pad_rows_bf16(const float* __restrict__ in, unsigned short* __restrict__ out,
                                int inrows, int outrows, int cols) {
  int t = blockIdx.x * blockDim.x + threadIdx.x;
  if (t >= outrows * cols) return;
  int r = t / cols, c = t - r * cols;
  float v = (r < inrows) ? in[(size_t)r * cols + c] : 0.0f;
  out[t] = f2bf(v);
}

// ---------------- WMMA GEMM: D[n,128] = A[n,KP] * B[KP,128] ----------------
// 256 threads = 8 wave32s; block tile 128 rows x 128 cols; wave owns 16 rows.
// W staged in LDS via CDNA5 async-to-LDS (ASYNCcnt), padded stride vs bank conflicts.
template <int KP>
__global__ __launch_bounds__(256) void k_gemm_wmma_bf16(
    const unsigned short* __restrict__ A,   // bf16 [n, KP]
    const unsigned short* __restrict__ B,   // bf16 [KP, 128]
    float* __restrict__ D,                  // f32 [n, 128]
    int n)
{
  constexpr int LDB = HID + 8;              // 136 elems = 272 B row stride
  __shared__ unsigned short Bs[KP * LDB];

  const int tid = threadIdx.x;

  // ---- cooperative stage of W into LDS ----
#pragma unroll
  for (int c = tid; c < KP * 8; c += 256) { // 8 x 16B chunks per 128-col row
    const int r = c >> 3, q = c & 7;
    const unsigned short* gp = B + r * HID + q * 16;
    unsigned short* lp = &Bs[r * LDB + q * 16];
#if HAVE_ASYNC_LDS
    __builtin_amdgcn_global_load_async_to_lds_b128((gptr_v4i)gp, (lptr_v4i)lp, 0, 0);
#else
    *(uint4*)lp = *(const uint4*)gp;
#endif
  }
#if HAVE_ASYNC_LDS
  __builtin_amdgcn_s_wait_asynccnt(0);
#endif
  __syncthreads();

  const int wave = tid >> 5;
  const int lane = tid & 31;
  const int m    = lane & 15;
  const int hi   = lane >> 4;
  const int rowBase = blockIdx.x * 128 + wave * 16;
  const int arow = rowBase + m;
  const bool inb = arow < n;

  v8f acc[8] = {};

#pragma unroll
  for (int k0 = 0; k0 < KP; k0 += 32) {
    // A fragment (global, streaming): lane = row M (ISA 16-bit A 16x32 layout)
    FragU fa;
    if (inb) {
      const unsigned short* p = A + (size_t)arow * KP + k0 + hi * 8;
      fa.u[0] = *(const uint4*)(p);
      fa.u[1] = *(const uint4*)(p + 16);
    } else {
      fa.u[0] = make_uint4(0u, 0u, 0u, 0u);
      fa.u[1] = make_uint4(0u, 0u, 0u, 0u);
    }
    // B fragments (LDS): lane = N column (m), elements walk K: k0+8*hi+s (s<8), k0+16+8*hi+(s-8) (s>=8).
    // Preload all 8 tiles, then issue the 8 WMMAs back-to-back.
    const unsigned short* bl = &Bs[(k0 + hi * 8) * LDB + m];
    FragU fb[8];
#pragma unroll
    for (int ct = 0; ct < 8; ++ct) {
      unsigned short* e = (unsigned short*)&fb[ct];
#pragma unroll
      for (int s = 0; s < 8; ++s) {
        e[s]     = bl[s * LDB + ct * 16];
        e[8 + s] = bl[(16 + s) * LDB + ct * 16];
      }
    }
#pragma unroll
    for (int ct = 0; ct < 8; ++ct) {
      acc[ct] = __builtin_amdgcn_wmma_f32_16x16x32_bf16(
          false, fa.v, false, fb[ct].v, (short)0, acc[ct], false, false);
    }
  }

  // C/D layout: VGPR v, lanes0-15: M=v, lanes16-31: M=v+8; N = lane&15
#pragma unroll
  for (int ct = 0; ct < 8; ++ct) {
    const int col = ct * 16 + m;
#pragma unroll
    for (int v = 0; v < 8; ++v) {
      const int r = rowBase + hi * 8 + v;
      if (r < n) D[(size_t)r * HID + col] = acc[ct][v];
    }
  }
}

// ---------------- seed: agg = h * dis^2 + bias ----------------
__global__ void k_seed(const float* __restrict__ h, const float* __restrict__ dis,
                       const float* __restrict__ bias, float* __restrict__ agg, int n) {
  int t = blockIdx.x * blockDim.x + threadIdx.x;
  if (t >= n * HID) return;
  int i = t >> 7, f = t & (HID - 1);
  float s = dis[i];
  agg[t] = h[t] * s * s + bias[f];
}

// ---------------- edge scatter: agg[dst] += h[src] * dis[src]*dis[dst] ----------------
__global__ __launch_bounds__(256) void k_scatter(
    const float* __restrict__ h, const int* __restrict__ src, const int* __restrict__ dst,
    const float* __restrict__ dis, float* __restrict__ agg, int e)
{
  unsigned t = blockIdx.x * blockDim.x + threadIdx.x;
  int ei = (int)(t >> 5);
  int q  = (int)(t & 31);
  if (ei >= e) return;
  int s = src[ei];
  int d = dst[ei];
  if (ei + 1 < e) __builtin_prefetch(h + (size_t)src[ei + 1] * HID, 0, 1);
  float nrm = dis[s] * dis[d];
  float4 v = *(const float4*)(h + (size_t)s * HID + q * 4);
  float* o = agg + (size_t)d * HID + q * 4;
  atomicAdd(o + 0, v.x * nrm);
  atomicAdd(o + 1, v.y * nrm);
  atomicAdd(o + 2, v.z * nrm);
  atomicAdd(o + 3, v.w * nrm);
}

// ---------------- relu variants ----------------
__global__ void k_relu_bf16(const float* __restrict__ agg, unsigned short* __restrict__ out, int n) {
  int t = blockIdx.x * blockDim.x + threadIdx.x;
  if (t < n * HID) out[t] = f2bf(fmaxf(agg[t], 0.0f));
}
__global__ void k_relu_f32(const float* __restrict__ agg, float* __restrict__ out, int n) {
  int t = blockIdx.x * blockDim.x + threadIdx.x;
  if (t < n * HID) out[t] = fmaxf(agg[t], 0.0f);
}

static inline size_t alignup(size_t x) { return (x + 255) & ~(size_t)255; }
static inline int cdiv(long long a, int b) { return (int)((a + b - 1) / b); }

extern "C" void kernel_launch(void* const* d_in, const int* in_sizes, int n_in,
                              void* d_out, int out_size, void* d_ws, size_t ws_size,
                              hipStream_t stream) {
  const float* x  = (const float*)d_in[0];
  const int*   ei = (const int*)d_in[1];
  const float* W1 = (const float*)d_in[2];
  const float* b1 = (const float*)d_in[3];
  const float* W2 = (const float*)d_in[4];
  const float* b2 = (const float*)d_in[5];
  const int* src = ei;
  const int* dst = ei + N_EDGES;

  char* ws = (char*)d_ws;
  size_t o = 0;
  float* deg = (float*)(ws + o);            o += alignup((size_t)N_NODES * 4);
  float* dis = (float*)(ws + o);            o += alignup((size_t)N_NODES * 4);
  unsigned short* xb = (unsigned short*)(ws + o); o += alignup((size_t)N_NODES * KP1 * 2);
  unsigned short* wb = (unsigned short*)(ws + o); o += alignup((size_t)HID * HID * 2);
  float* h   = (float*)(ws + o);            o += alignup((size_t)N_NODES * HID * 4);
  float* agg = (float*)(ws + o);            o += alignup((size_t)N_NODES * HID * 4);
  unsigned short* hb = (unsigned short*)(ws + o); o += alignup((size_t)N_NODES * HID * 2);
  (void)ws_size; (void)n_in; (void)in_sizes;

  const int T = 256;
  const int gemm_blocks = cdiv(N_NODES, 128);

  // degree + normalization
  k_init_deg<<<cdiv(N_NODES, T), T, 0, stream>>>(deg, N_NODES);
  k_deg_accum<<<cdiv(N_EDGES, T), T, 0, stream>>>(dst, deg, N_EDGES);
  k_rsqrt<<<cdiv(N_NODES, T), T, 0, stream>>>(deg, dis, N_NODES);

  // ---- layer 1 ----
  k_pad_cols_bf16<<<cdiv((long long)N_NODES * KP1, T), T, 0, stream>>>(x, xb, N_NODES, IN_DIM, KP1);
  k_pad_rows_bf16<<<cdiv((long long)KP1 * HID, T), T, 0, stream>>>(W1, wb, IN_DIM, KP1, HID);
  k_gemm_wmma_bf16<KP1><<<gemm_blocks, T, 0, stream>>>(xb, wb, h, N_NODES);
  k_seed<<<cdiv((long long)N_NODES * HID, T), T, 0, stream>>>(h, dis, b1, agg, N_NODES);
  k_scatter<<<cdiv((long long)N_EDGES * 32, T), T, 0, stream>>>(h, src, dst, dis, agg, N_EDGES);
  k_relu_bf16<<<cdiv((long long)N_NODES * HID, T), T, 0, stream>>>(agg, hb, N_NODES);

  // ---- layer 2 ----
  k_pad_rows_bf16<<<cdiv((long long)HID * HID, T), T, 0, stream>>>(W2, wb, HID, HID, HID);
  k_gemm_wmma_bf16<HID><<<gemm_blocks, T, 0, stream>>>(hb, wb, h, N_NODES);
  k_seed<<<cdiv((long long)N_NODES * HID, T), T, 0, stream>>>(h, dis, b2, agg, N_NODES);
  k_scatter<<<cdiv((long long)N_EDGES * 32, T), T, 0, stream>>>(h, src, dst, dis, agg, N_EDGES);
  k_relu_f32<<<cdiv((long long)N_NODES * HID, T), T, 0, stream>>>(agg, (float*)d_out, N_NODES);

  // ---- pass edge_index through as second tuple output ----
  long long hdr = (long long)N_NODES * HID;
  if ((long long)out_size > hdr) {
    size_t tail_bytes = (size_t)((long long)out_size - hdr) * 4;
    (void)hipMemcpyAsync((char*)d_out + (size_t)hdr * 4, d_in[1], tail_bytes,
                         hipMemcpyDeviceToDevice, stream);
  }
}
